// Sam2VideoRoPEAttention_24850680775318
// MI455X (gfx1250) — matrix operand order repaired and checked by reading, not compiled, not run on hardware
//
#include <hip/hip_runtime.h>

typedef __attribute__((ext_vector_type(16))) __bf16 v16bf;
typedef __attribute__((ext_vector_type(8)))  __bf16 v8bf;
typedef __attribute__((ext_vector_type(8)))  float  v8f;

#define QS   4096
#define KSEQ 16448
#define HID  256
#define KVIN 64
#define SCALE_F 0.0625f
#define KSPLIT 8

// ---------- WMMA helpers ----------
__device__ __forceinline__ v8f wmma_bf16(v16bf a, v16bf b, v8f c) {
  return __builtin_amdgcn_wmma_f32_16x16x32_bf16(false, a, false, b, (short)0, c,
                                                 false, false);
}

__device__ __forceinline__ v16bf ld16(const __bf16* p) {
  v8bf lo = *(const v8bf*)p;
  v8bf hi = *(const v8bf*)(p + 8);
  return __builtin_shufflevector(lo, hi, 0,1,2,3,4,5,6,7,8,9,10,11,12,13,14,15);
}
__device__ __forceinline__ v16bf ld8x2(const __bf16* p0, const __bf16* p1) {
  v8bf lo = *(const v8bf*)p0;
  v8bf hi = *(const v8bf*)p1;
  return __builtin_shufflevector(lo, hi, 0,1,2,3,4,5,6,7,8,9,10,11,12,13,14,15);
}

// A-fragment loaders: lane holds row m, k-runs [hs*8..+7] and [16+hs*8..+7]
__device__ __forceinline__ v16bf loadA(const float* arow, int hs) {
  const v8f lo = *(const v8f*)(arow + hs * 8);
  const v8f hi = *(const v8f*)(arow + 16 + hs * 8);
  v16bf a;
#pragma unroll
  for (int j = 0; j < 8; ++j) { a[j] = (__bf16)lo[j]; a[8 + j] = (__bf16)hi[j]; }
  return a;
}
__device__ __forceinline__ v16bf loadA(const __bf16* arow, int hs) {
  return ld8x2(arow + hs * 8, arow + 16 + hs * 8);
}

__device__ __forceinline__ float rmax16(float x) {
  x = fmaxf(x, __shfl_xor(x, 1, 32));
  x = fmaxf(x, __shfl_xor(x, 2, 32));
  x = fmaxf(x, __shfl_xor(x, 4, 32));
  x = fmaxf(x, __shfl_xor(x, 8, 32));
  return x;
}
__device__ __forceinline__ float rsum16(float x) {
  x += __shfl_xor(x, 1, 32);
  x += __shfl_xor(x, 2, 32);
  x += __shfl_xor(x, 4, 32);
  x += __shfl_xor(x, 8, 32);
  return x;
}

// ---------- Weight transpose to bf16: Wt[N][K] = (bf16)W[K][N] ----------
__global__ void wtrans_kernel(const float* __restrict__ W, __bf16* __restrict__ Wt,
                              int K, int N) {
  const size_t idx = (size_t)blockIdx.x * blockDim.x + threadIdx.x; // K*N
  const int k = (int)(idx / N);
  const int n = (int)(idx % N);
  Wt[(size_t)n * K + k] = (__bf16)W[idx];
}

// ---------- bf16-WMMA GEMM: C[M,N] = A[M,K] @ Bt[N,K]^T + bias ----------
template <typename AT>
__global__ __launch_bounds__(128, 1) void gemm_wmma(
    const AT* __restrict__ A, const __bf16* __restrict__ Bt,
    const float* __restrict__ bias, float* __restrict__ C, int K, int N) {
  const int lane = threadIdx.x & 31;
  const int wv   = threadIdx.x >> 5;
  const int lm   = lane & 15;
  const int hs   = lane >> 4;
  const int NT   = N >> 4;
  const int tile = blockIdx.x * (blockDim.x >> 5) + wv;
  const int m0   = (tile / NT) << 4;
  const int n0   = (tile % NT) << 4;

  v8f c;
#pragma unroll
  for (int j = 0; j < 8; ++j) c[j] = 0.0f;

  for (int kc = 0; kc < K; kc += 32) {
    const v16bf a = loadA(A + (size_t)(m0 + lm) * K + kc, hs);
    const v16bf b = ld8x2(Bt + (size_t)(n0 + lm) * K + kc + hs * 8,
                          Bt + (size_t)(n0 + lm) * K + kc + 16 + hs * 8);
    c = wmma_bf16(a, b, c);
  }
  const float bb = bias[n0 + lm];
#pragma unroll
  for (int r = 0; r < 8; ++r)
    C[(size_t)(m0 + r + hs * 8) * N + n0 + lm] = c[r] + bb;
}

// ---------- RoPE on q (folds SCALE), f32 -> bf16 ----------
__global__ void rope_q_kernel(const float* __restrict__ qt, const float* __restrict__ cs,
                              const float* __restrict__ sn, __bf16* __restrict__ Qb) {
  const size_t idx = (size_t)blockIdx.x * blockDim.x + threadIdx.x; // QS*HID
  const int col = (int)(idx & (HID - 1));
  const float v = qt[idx];
  const float partner = qt[idx ^ 1];
  const float rot = (col & 1) ? partner : -partner;
  Qb[idx] = (__bf16)((v * cs[idx] + rot * sn[idx]) * SCALE_F);
}

// ---------- RoPE on k rows < n_rot, f32 -> bf16 ----------
__global__ void rope_k_kernel(const float* __restrict__ kt, const float* __restrict__ cs,
                              const float* __restrict__ sn, const int* __restrict__ excl,
                              __bf16* __restrict__ Kb) {
  const size_t idx = (size_t)blockIdx.x * blockDim.x + threadIdx.x; // KSEQ*HID
  const int row = (int)(idx >> 8);
  const int col = (int)(idx & (HID - 1));
  const int n_rot = KSEQ - *excl;
  const float v = kt[idx];
  float o = v;
  if (row < n_rot) {
    int rep = n_rot / QS;
    if (rep < 1) rep = 1;
    const int crow = row / rep;
    const float partner = kt[idx ^ 1];
    const float rot = (col & 1) ? partner : -partner;
    o = v * cs[(size_t)crow * HID + col] + rot * sn[(size_t)crow * HID + col];
  }
  Kb[idx] = (__bf16)o;
}

// ---------- V transpose: Vt[d][k] = v_proj[k][d], f32 -> bf16 ----------
__global__ void vtrans_kernel(const float* __restrict__ vt, __bf16* __restrict__ Vt) {
  const size_t idx = (size_t)blockIdx.x * blockDim.x + threadIdx.x; // KSEQ*HID
  const int k = (int)(idx >> 8);
  const int d = (int)(idx & (HID - 1));
  Vt[(size_t)d * KSEQ + k] = (__bf16)vt[idx];
}

// ---------- Flash attention, key-split: one wave = 16 q rows x 256 d, 1/8 keys ----
// Waves within a block share the SAME key split -> identical K/V streams (L0 reuse).
__global__ __launch_bounds__(128, 1) void attn_kernel(
    const __bf16* __restrict__ Qb, const __bf16* __restrict__ Kb,
    const __bf16* __restrict__ Vt, float* __restrict__ Opart,
    float* __restrict__ Mp, float* __restrict__ Lp) {
  __shared__ __align__(16) __bf16 plds[4][16][32];
  const int lane = threadIdx.x & 31;
  const int wv   = threadIdx.x >> 5;
  const int lm   = lane & 15;
  const int hs   = lane >> 4;
  const int wgl  = blockIdx.x * 4 + wv;    // 0..2047
  const int spl  = wgl >> 8;               // key split 0..7 (uniform per block)
  const int qtl  = wgl & 255;              // query tile 0..255
  const int m0   = qtl << 4;

  // chunk range for this split: 514 chunks -> 65,65,64,64,64,64,64,64
  const int extra = (spl < 2) ? spl : 2;
  const int c0 = spl * 64 + extra;
  const int c1 = c0 + 64 + ((spl < 2) ? 1 : 0);

  // Preload Q A-fragments (kept resident via VGPR-MSB register file)
  const __bf16* qrow = Qb + (size_t)(m0 + lm) * HID;
  v16bf qf[8];
#pragma unroll
  for (int t = 0; t < 8; ++t)
    qf[t] = ld8x2(qrow + t * 32 + hs * 8, qrow + t * 32 + 16 + hs * 8);

  v8f o[16];
#pragma unroll
  for (int t = 0; t < 16; ++t)
#pragma unroll
    for (int j = 0; j < 8; ++j) o[t][j] = 0.0f;

  float mrun[8], lrun[8];
#pragma unroll
  for (int r = 0; r < 8; ++r) { mrun[r] = -1e30f; lrun[r] = 0.0f; }

  for (int kb = c0 * 32; kb < c1 * 32; kb += 32) {
    // ---- batch-load all 16 K fragments for this chunk (one long clause) ----
    const __bf16* kr0 = Kb + (size_t)(kb + lm) * HID + hs * 8;
    const __bf16* kr1 = Kb + (size_t)(kb + 16 + lm) * HID + hs * 8;
    v16bf kf0[8], kf1[8];
#pragma unroll
    for (int t = 0; t < 8; ++t) kf0[t] = ld8x2(kr0 + t * 32, kr0 + t * 32 + 16);
#pragma unroll
    for (int t = 0; t < 8; ++t) kf1[t] = ld8x2(kr1 + t * 32, kr1 + t * 32 + 16);

    // ---- S = Q.K^T : 16 back-to-back WMMAs ----
    v8f s0, s1;
#pragma unroll
    for (int j = 0; j < 8; ++j) { s0[j] = 0.0f; s1[j] = 0.0f; }
#pragma unroll
    for (int t = 0; t < 8; ++t) s0 = wmma_bf16(qf[t], kf0[t], s0);
#pragma unroll
    for (int t = 0; t < 8; ++t) s1 = wmma_bf16(qf[t], kf1[t], s1);

    // ---- prefetch V fragments NOW; they overlap the softmax VALU phase ----
    v16bf vf[16];
#pragma unroll
    for (int t = 0; t < 16; ++t)
      vf[t] = ld8x2(Vt + (size_t)(t * 16 + lm) * KSEQ + kb + hs * 8,
                    Vt + (size_t)(t * 16 + lm) * KSEQ + kb + 16 + hs * 8);

    // ---- online softmax (row r lives in one 16-lane group) ----
#pragma unroll
    for (int r = 0; r < 8; ++r) {
      const float cmax = rmax16(fmaxf(s0[r], s1[r]));
      const float mnew = fmaxf(mrun[r], cmax);
      const float corr = __expf(mrun[r] - mnew);
      const float p0 = __expf(s0[r] - mnew);
      const float p1 = __expf(s1[r] - mnew);
      const float ps = rsum16(p0 + p1);
      lrun[r] = lrun[r] * corr + ps;
      mrun[r] = mnew;
#pragma unroll
      for (int t = 0; t < 16; ++t) o[t][r] *= corr;
      plds[wv][r + hs * 8][lm]      = (__bf16)p0;
      plds[wv][r + hs * 8][16 + lm] = (__bf16)p1;
    }
    asm volatile("s_wait_dscnt 0" ::: "memory"); // P stores visible to whole wave
    const v16bf pf = ld8x2(&plds[wv][lm][hs * 8], &plds[wv][lm][16 + hs * 8]);
    asm volatile("s_wait_dscnt 0" ::: "memory"); // P reads done before next chunk

    // ---- O += P.V : 16 back-to-back WMMAs on preloaded V fragments ----
#pragma unroll
    for (int t = 0; t < 16; ++t) o[t] = wmma_bf16(pf, vf[t], o[t]);
  }

  // Store unnormalized partial O plus per-row (m, l)
  const size_t prow = (size_t)(qtl * KSPLIT + spl) * 16;
#pragma unroll
  for (int r = 0; r < 8; ++r) {
    const size_t pm = prow + r + hs * 8;
#pragma unroll
    for (int t = 0; t < 16; ++t)
      Opart[pm * HID + t * 16 + lm] = o[t][r];
  }
  if (lm == 0) {
#pragma unroll
    for (int r = 0; r < 8; ++r) {
      Mp[prow + r + hs * 8] = mrun[r];
      Lp[prow + r + hs * 8] = lrun[r];
    }
  }
}

// ---------- Merge key-split partials (log-sum-exp combine) -> bf16 x ----------
__global__ void merge_kernel(const float* __restrict__ Opart, const float* __restrict__ Mp,
                             const float* __restrict__ Lp, __bf16* __restrict__ Xb) {
  const size_t idx = (size_t)blockIdx.x * blockDim.x + threadIdx.x; // QS*HID
  const int m = (int)(idx >> 8);
  const int d = (int)(idx & (HID - 1));
  const int qtl = m >> 4, r = m & 15;
  const size_t base = (size_t)qtl * KSPLIT * 16 + r;
  float M = -1e30f;
#pragma unroll
  for (int s = 0; s < KSPLIT; ++s) M = fmaxf(M, Mp[base + s * 16]);
  float L = 0.0f, acc = 0.0f;
#pragma unroll
  for (int s = 0; s < KSPLIT; ++s) {
    const float w = __expf(Mp[base + s * 16] - M);
    L += w * Lp[base + s * 16];
    acc += w * Opart[(base + s * 16) * HID + d];
  }
  Xb[idx] = (__bf16)(acc / L);
}

extern "C" void kernel_launch(void* const* d_in, const int* in_sizes, int n_in,
                              void* d_out, int out_size, void* d_ws, size_t ws_size,
                              hipStream_t stream) {
  (void)in_sizes; (void)n_in; (void)out_size; (void)ws_size;
  const float* query = (const float*)d_in[0];
  const float* key   = (const float*)d_in[1];
  const float* value = (const float*)d_in[2];
  const float* cosb  = (const float*)d_in[3];
  const float* sinb  = (const float*)d_in[4];
  const float* wq    = (const float*)d_in[5];
  const float* bq    = (const float*)d_in[6];
  const float* wk    = (const float*)d_in[7];
  const float* bk    = (const float*)d_in[8];
  const float* wv    = (const float*)d_in[9];
  const float* bv    = (const float*)d_in[10];
  const float* wo    = (const float*)d_in[11];
  const float* bo    = (const float*)d_in[12];
  const int*   excl  = (const int*)d_in[13];
  float* out = (float*)d_out;

  char* ws = (char*)d_ws;
  size_t off = 0;
  const size_t qtmp_off = off; off += (size_t)QS   * HID * 4;
  const size_t ktmp_off = off; off += (size_t)KSEQ * HID * 4;
  float*  vtmp = (float*)(ws + off);  off += (size_t)KSEQ * HID * 4;
  __bf16* Qb   = (__bf16*)(ws + off); off += (size_t)QS   * HID * 2;
  __bf16* Kb   = (__bf16*)(ws + off); off += (size_t)KSEQ * HID * 2;
  __bf16* Vt   = (__bf16*)(ws + off); off += (size_t)KSEQ * HID * 2;
  __bf16* Xb   = (__bf16*)(ws + off); off += (size_t)QS   * HID * 2;
  __bf16* Wqt  = (__bf16*)(ws + off); off += (size_t)HID  * HID * 2;
  __bf16* Wkt  = (__bf16*)(ws + off); off += (size_t)HID  * KVIN * 2;
  __bf16* Wvt  = (__bf16*)(ws + off); off += (size_t)HID  * KVIN * 2;
  __bf16* Wot  = (__bf16*)(ws + off); off += (size_t)HID  * HID * 2; // ~59.2 MB total

  // Attention partials alias dead f32 temporaries:
  //   Opart (32 MB) overlays ktmp+vtmp (33.7 MB, dead after RoPE/vtrans)
  //   Mp/Lp overlay qtmp (dead after rope_q)
  float* Opart = (float*)(ws + ktmp_off);
  float* Mpart = (float*)(ws + qtmp_off);
  float* Lpart = (float*)(ws + qtmp_off + (size_t)(QS / 16) * KSPLIT * 16 * 4);

  const dim3 blk(128); // 4 waves per block (wave32)

  // Weight transposes (bf16 [N][K]) for contiguous B-fragment loads
  wtrans_kernel<<<dim3(HID * HID / 256), dim3(256), 0, stream>>>(wq, Wqt, HID, HID);
  wtrans_kernel<<<dim3(KVIN * HID / 256), dim3(256), 0, stream>>>(wk, Wkt, KVIN, HID);
  wtrans_kernel<<<dim3(KVIN * HID / 256), dim3(256), 0, stream>>>(wv, Wvt, KVIN, HID);
  wtrans_kernel<<<dim3(HID * HID / 256), dim3(256), 0, stream>>>(wo, Wot, HID, HID);

  // Projections (WMMA bf16, f32 accumulate, f32 out)
  gemm_wmma<float><<<dim3((QS / 16) * (HID / 16) / 4), blk, 0, stream>>>(
      query, Wqt, bq, (float*)(ws + qtmp_off), HID, HID);
  gemm_wmma<float><<<dim3((KSEQ / 16) * (HID / 16) / 4), blk, 0, stream>>>(
      key, Wkt, bk, (float*)(ws + ktmp_off), KVIN, HID);
  gemm_wmma<float><<<dim3((KSEQ / 16) * (HID / 16) / 4), blk, 0, stream>>>(
      value, Wvt, bv, vtmp, KVIN, HID);

  // RoPE + layout conversion to bf16
  rope_q_kernel<<<dim3(QS * HID / 256), dim3(256), 0, stream>>>(
      (const float*)(ws + qtmp_off), cosb, sinb, Qb);
  rope_k_kernel<<<dim3(KSEQ * HID / 256), dim3(256), 0, stream>>>(
      (const float*)(ws + ktmp_off), cosb, sinb, excl, Kb);
  vtrans_kernel<<<dim3(KSEQ * HID / 256), dim3(256), 0, stream>>>(vtmp, Vt);

  // Flash attention: 256 q-tiles x 8 key splits = 2048 waves
  attn_kernel<<<dim3(QS / 16 * KSPLIT / 4), blk, 0, stream>>>(Qb, Kb, Vt,
                                                              Opart, Mpart, Lpart);
  merge_kernel<<<dim3(QS * HID / 256), dim3(256), 0, stream>>>(Opart, Mpart, Lpart, Xb);

  // Output projection -> f32 d_out
  gemm_wmma<__bf16><<<dim3((QS / 16) * (HID / 16) / 4), blk, 0, stream>>>(
      Xb, Wot, bo, out, HID, HID);
}
